// HiPPO_11304353923243
// MI455X (gfx1250) — matrix operand order repaired and checked by reading, not compiled, not run on hardware
//
#include <hip/hip_runtime.h>
#include <hip/hip_bf16.h>
#include <stdint.h>

// ---------------------------------------------------------------------------
// HiPPO-LegT scan refactored into batched WMMA GEMMs for gfx1250 (MI455X).
//   N=256 state, B=64 batch, L=4096 steps, chunk T=64 -> 64 chunks.
//   out[k*64+j] = V[k] (64x320) @ G[j] (320x256),  f16 WMMA, f32 accum.
// ---------------------------------------------------------------------------

typedef _Float16 f16;
typedef __attribute__((ext_vector_type(16))) _Float16 v16h;
typedef __attribute__((ext_vector_type(8)))  _Float16 v8h;
typedef __attribute__((ext_vector_type(8)))  float    v8f;

#define N_   256
#define T_   64
#define KC_  64      // number of chunks
#define AUG_ 320     // N_ + T_
#define BAT_ 64

// ---------------- setup kernel 1: Mpow[0] = A^T, zero carry[0] --------------
__global__ __launch_bounds__(256) void k_mpow0(const float* __restrict__ A,
                                               float* __restrict__ Mpow0,
                                               float* __restrict__ carry0) {
  int m = blockIdx.x, n = threadIdx.x;
  Mpow0[m * N_ + n] = A[n * N_ + m];          // M = A^T
  if (m < BAT_) carry0[m * N_ + n] = 0.0f;    // c_carry[0] = 0
}

// ---------------- setup kernel 2: Kv[j] = A^j B (63 sequential matvecs) -----
__global__ __launch_bounds__(256) void k_kv(const float* __restrict__ A,
                                            const float* __restrict__ Bvec,
                                            float* __restrict__ Kv) {
  __shared__ float prev[N_];
  int n = threadIdx.x;
  float v = Bvec[n];
  Kv[n] = v;
  prev[n] = v;
  __syncthreads();
  for (int j = 1; j < T_; ++j) {
    float acc = 0.0f;
    for (int m = 0; m < N_; ++m) acc += A[n * N_ + m] * prev[m];
    __syncthreads();
    prev[n] = acc;
    Kv[j * N_ + n] = acc;
    __syncthreads();
  }
}

// ---------------- setup kernel 3: power doubling  C_i = Aconst @ B_i --------
// Produces Mpow[s..s+cnt-1] = Mpow[s-1] @ Mpow[0..cnt-1]  (f32, 256x256x256)
__global__ __launch_bounds__(256) void k_matmul256(const float* __restrict__ Am,
                                                   const float* __restrict__ Bbase,
                                                   float* __restrict__ Cbase) {
  int i = blockIdx.z;
  const float* Bm = Bbase + (size_t)i * N_ * N_;
  float*       Cm = Cbase + (size_t)i * N_ * N_;
  __shared__ float As[16][16];
  __shared__ float Bs[16][17];
  int tx = threadIdx.x, ty = threadIdx.y;
  int row = blockIdx.y * 16 + ty;
  int col = blockIdx.x * 16 + tx;
  float acc = 0.0f;
  for (int t = 0; t < N_; t += 16) {
    As[ty][tx] = Am[row * N_ + t + tx];
    Bs[ty][tx] = Bm[(t + ty) * N_ + col];
    __syncthreads();
#pragma unroll
    for (int kk = 0; kk < 16; ++kk) acc += As[ty][kk] * Bs[kk][tx];
    __syncthreads();
  }
  Cm[row * N_ + col] = acc;
}

// ---------------- setup kernel 4: ll[k][b][n] = sum_i u[kT+i,b]*Kv[T-1-i][n] -
__global__ __launch_bounds__(256) void k_local_last(const float* __restrict__ u,
                                                    const float* __restrict__ Kv,
                                                    float* __restrict__ ll) {
  int k = blockIdx.x;
  __shared__ float ul[T_ * BAT_];
  for (int idx = threadIdx.x; idx < T_ * BAT_; idx += 256)
    ul[idx] = u[(size_t)k * T_ * BAT_ + idx];       // u[(k*64+i)*64+b]
  __syncthreads();
  int n = threadIdx.x;
  float acc[BAT_];
#pragma unroll
  for (int b = 0; b < BAT_; ++b) acc[b] = 0.0f;
  for (int i = 0; i < T_; ++i) {
    float kvv = Kv[(T_ - 1 - i) * N_ + n];
#pragma unroll
    for (int b = 0; b < BAT_; ++b) acc[b] += ul[i * BAT_ + b] * kvv;
  }
#pragma unroll
  for (int b = 0; b < BAT_; ++b) ll[((size_t)k * BAT_ + b) * N_ + n] = acc[b];
}

// ---------------- setup kernel 5: one carry-scan step (f32, precision) ------
// Cout(64x256) = Cprev @ M64(256x256) + llk
__global__ __launch_bounds__(256) void k_carry_step(const float* __restrict__ Cprev,
                                                    const float* __restrict__ M64,
                                                    const float* __restrict__ llk,
                                                    float* __restrict__ Cout) {
  __shared__ float As[16][16];
  __shared__ float Bs[16][17];
  int tx = threadIdx.x, ty = threadIdx.y;
  int row = blockIdx.y * 16 + ty;                   // 0..63
  int col = blockIdx.x * 16 + tx;                   // 0..255
  float acc = 0.0f;
  for (int t = 0; t < N_; t += 16) {
    As[ty][tx] = Cprev[row * N_ + t + tx];
    Bs[ty][tx] = M64[(t + ty) * N_ + col];
    __syncthreads();
#pragma unroll
    for (int kk = 0; kk < 16; ++kk) acc += As[ty][kk] * Bs[kk][tx];
    __syncthreads();
  }
  Cout[row * N_ + col] = acc + llk[row * N_ + col];
}

// ---------------- pack kernel: Gt[j][n][kk] (f16), kk = [Mpow | W] -----------
__global__ __launch_bounds__(320) void k_build_gt(const float* __restrict__ Mpow,
                                                  const float* __restrict__ Kv,
                                                  f16* __restrict__ Gt) {
  int j = blockIdx.x, n = blockIdx.y, kk = threadIdx.x;
  float v;
  if (kk < N_) {
    v = Mpow[((size_t)j * N_ + kk) * N_ + n];       // Mpow[j][kk][n]
  } else {
    int i = kk - N_;
    v = (i <= j) ? Kv[(j - i) * N_ + n] : 0.0f;     // W[j][i][n]
  }
  Gt[((size_t)j * N_ + n) * AUG_ + kk] = (f16)v;
}

// ---------------- pack kernel: V[k][b][kk] = [carry[k] | u_chunk] (f16) ------
__global__ __launch_bounds__(320) void k_build_v(const float* __restrict__ carry,
                                                 const float* __restrict__ u,
                                                 f16* __restrict__ V) {
  int k = blockIdx.x, b = blockIdx.y, kk = threadIdx.x;
  float v = (kk < N_) ? carry[((size_t)k * BAT_ + b) * N_ + kk]
                      : u[((size_t)k * T_ + (kk - N_)) * BAT_ + b];
  V[((size_t)k * BAT_ + b) * AUG_ + kk] = (f16)v;
}

// ---------------- main phase-B kernel: out[k,j] = V[k] @ G[j] (WMMA f16) -----
__global__ __launch_bounds__(256) void k_phaseB(const f16* __restrict__ V,
                                                const f16* __restrict__ Gt,
                                                float* __restrict__ out) {
  const int k = blockIdx.x;
  const int j = blockIdx.y;

  __shared__ __align__(32) f16 Vlds[BAT_ * AUG_];   // 40 KB
  const f16* Vg = V + (size_t)k * BAT_ * AUG_;

#if defined(__gfx1250__) && __has_builtin(__builtin_amdgcn_tensor_load_to_lds)
  // --- Tensor Data Mover: DMA the 64x320 f16 tile into LDS -------------------
  if (threadIdx.x < 32) {
    typedef unsigned int u32x4 __attribute__((ext_vector_type(4)));
    typedef int          i32x8 __attribute__((ext_vector_type(8)));
    typedef int          i32x4 __attribute__((ext_vector_type(4)));
    unsigned long long ga = (unsigned long long)(uintptr_t)Vg;
    unsigned lds_off = (unsigned)(uintptr_t)(void*)Vlds;   // addr[31:0] = LDS offset
    // D# group 0: count=1 | lds_addr | global_addr[56:0] | type=2
    u32x4 g0 = { 1u, lds_off, (unsigned)(ga & 0xffffffffu),
                 (unsigned)((ga >> 32) & 0x01ffffffu) | (2u << 30) };
    // D# group 1: data_size=2B(code1), tensor_dim0=320, tensor_dim1=64,
    //             tile_dim0=320, tile_dim1=64, tensor_dim0_stride=320
    i32x8 g1 = { (int)(1u << 16),
                 (int)(320u << 16),
                 (int)((64u << 16) | 0u),
                 (int)(320u << 16),
                 (int)64u,
                 (int)320u,
                 0, 0 };
    i32x4 gz4 = { 0, 0, 0, 0 };
    i32x8 gz8 = { 0, 0, 0, 0, 0, 0, 0, 0 };
    // amdgpu-toolchain (clang-23) 6-arg form:
    // (g0: u32x4, g1: i32x8, g2: i32x4, g3: i32x4, extra: i32x8, cpol: i32)
    __builtin_amdgcn_tensor_load_to_lds(g0, g1, gz4, gz4, gz8, 0);
    __builtin_amdgcn_s_wait_tensorcnt(0);
  }
#else
  {
    const unsigned* src = (const unsigned*)Vg;
    unsigned* dst = (unsigned*)Vlds;
    for (int idx = threadIdx.x; idx < BAT_ * AUG_ / 2; idx += 256)
      dst[idx] = src[idx];
  }
#endif
  __syncthreads();

  const int wid  = threadIdx.x >> 5;
  const int lane = threadIdx.x & 31;
  const int half = lane >> 4;
  const int lmod = lane & 15;
  const int mBase = (wid & 1) * 32;        // rows (batch)   : 2 M-tiles
  const int nBase = (wid >> 1) * 64;       // cols (state)   : 4 N-tiles

  v8f acc[2][4] = {};
  const f16* GtJ = Gt + (size_t)j * N_ * AUG_;

  for (int ks = 0; ks < AUG_; ks += 32) {
    // A fragments (16x32 f16) from LDS, per documented 16-bit A layout:
    // lane half h, v16h elems 0..7 -> K = ks + h*8 + (0..7), elems 8..15 -> +16
    v16h a[2];
#pragma unroll
    for (int mi = 0; mi < 2; ++mi) {
      const f16* p = &Vlds[(mBase + mi * 16 + lmod) * AUG_ + ks + half * 8];
      union { v16h v; v8h h[2]; } ua;
      ua.h[0] = *(const v8h*)p;
      ua.h[1] = *(const v8h*)(p + 16);
      a[mi] = ua.v;
    }
    // B fragments (32x16 f16) from global Gt (n-major), per 16-bit B layout:
    // lane holds column n = lmod, elems 0..7 -> K = ks + h*8 + (0..7), elems 8..15 -> +16
    v16h b[4];
#pragma unroll
    for (int ni = 0; ni < 4; ++ni) {
      const f16* p = GtJ + (size_t)(nBase + ni * 16 + lmod) * AUG_ + ks + half * 8;
      union { v16h v; v8h h[2]; } ub;
      ub.h[0] = *(const v8h*)p;
      ub.h[1] = *(const v8h*)(p + 16);
      b[ni] = ub.v;
    }
#pragma unroll
    for (int mi = 0; mi < 2; ++mi)
#pragma unroll
      for (int ni = 0; ni < 4; ++ni)
        acc[mi][ni] = __builtin_amdgcn_wmma_f32_16x16x32_f16(
            false, a[mi], false, b[ni], (short)0, acc[mi][ni], false, false);
  }

  // Store: C layout -> VGPR r holds row (r + half*8), col = lmod within tile.
  const size_t tbase = (size_t)((k * T_ + j) * BAT_) * N_;
#pragma unroll
  for (int mi = 0; mi < 2; ++mi)
#pragma unroll
    for (int ni = 0; ni < 4; ++ni)
#pragma unroll
      for (int r = 0; r < 8; ++r) {
        int row = mBase + mi * 16 + half * 8 + r;
        int col = nBase + ni * 16 + lmod;
        out[tbase + (size_t)row * N_ + col] = acc[mi][ni][r];
      }
}

// ---------------------------------------------------------------------------
extern "C" void kernel_launch(void* const* d_in, const int* in_sizes, int n_in,
                              void* d_out, int out_size, void* d_ws, size_t ws_size,
                              hipStream_t stream) {
  const float* u    = (const float*)d_in[0];   // (4096, 64)
  const float* A    = (const float*)d_in[1];   // (256, 256)
  const float* Bvec = (const float*)d_in[2];   // (256,)
  float* out = (float*)d_out;                  // (4096, 64, 256)

  char* ws = (char*)d_ws;
  size_t off = 0;
  auto take = [&](size_t bytes) -> void* {
    void* p = ws + off;
    off += (bytes + 255) & ~(size_t)255;
    return p;
  };
  float* Mpow  = (float*)take((size_t)KC_ * N_ * N_ * 4);   // 16 MB: M^(j+1)
  float* Kv    = (float*)take((size_t)T_ * N_ * 4);         // A^j B
  float* ll    = (float*)take((size_t)KC_ * BAT_ * N_ * 4); // chunk local tails
  float* carry = (float*)take((size_t)KC_ * BAT_ * N_ * 4); // chunk carries
  f16*   Gt    = (f16*)  take((size_t)T_ * N_ * AUG_ * 2);  // G[j]^T, f16
  f16*   V     = (f16*)  take((size_t)KC_ * BAT_ * AUG_ * 2);

  // 1) M = A^T, carry[0] = 0;  Kv[j] = A^j B
  k_mpow0<<<N_, 256, 0, stream>>>(A, Mpow, carry);
  k_kv<<<1, 256, 0, stream>>>(A, Bvec, Kv);

  // 2) log-depth power doubling: Mpow[s+i] = Mpow[s-1] @ Mpow[i]
  for (int s = 1; s < T_; s *= 2) {
    k_matmul256<<<dim3(16, 16, s), dim3(16, 16), 0, stream>>>(
        Mpow + (size_t)(s - 1) * N_ * N_, Mpow, Mpow + (size_t)s * N_ * N_);
  }

  // 3) per-chunk local tails, then sequential carry scan (f32, 63 tiny GEMMs)
  k_local_last<<<KC_, 256, 0, stream>>>(u, Kv, ll);
  for (int k = 1; k < KC_; ++k) {
    k_carry_step<<<dim3(16, 4), dim3(16, 16), 0, stream>>>(
        carry + (size_t)(k - 1) * BAT_ * N_,
        Mpow + (size_t)(T_ - 1) * N_ * N_,
        ll + (size_t)(k - 1) * BAT_ * N_,
        carry + (size_t)k * BAT_ * N_);
  }

  // 4) pack f16 operands
  k_build_gt<<<dim3(T_, N_), AUG_, 0, stream>>>(Mpow, Kv, Gt);
  k_build_v<<<dim3(KC_, BAT_), AUG_, 0, stream>>>(carry, u, V);

  // 5) phase B: 4096 independent 64x320x256 WMMA GEMMs -> all outputs
  k_phaseB<<<dim3(KC_, T_), 256, 0, stream>>>(V, Gt, out);
}
